// TransferNet_74698071212392
// MI455X (gfx1250) — matrix operand order repaired and checked by reading, not compiled, not run on hardware
//
#include <hip/hip_runtime.h>
#include <math.h>

typedef __attribute__((ext_vector_type(16))) _Float16 v16h;
typedef __attribute__((ext_vector_type(8)))  float    v8f;

#define NUM_ENT   100000
#define NUM_EDGES 200000
#define DIM       768
#define KCHUNKS   24      // 768 / 32
#define TM        64      // rows per block in the big GEMM

__device__ __forceinline__ float gelu_exact(float x) {
    return 0.5f * x * (1.0f + erff(x * 0.70710678118654752f));
}

// ---------------------------------------------------------------------------
// Kernel 1: small per-step MLPs + attention, build the swizzled f16 B matrix
// W1sw[kc][nt][lane][p] as WMMA-B (32x16 f16) fragments, contiguous k order (same as A):
//   lane = (n&15) + 16*(kk>>4),  p = kk&15,  kk = k&31, kc = k>>5, nt = t*16 + (n>>4)
// grid = 3 blocks (one per step), 256 threads
// ---------------------------------------------------------------------------
__global__ void prep_kernel(const float* __restrict__ q_emb,
                            const float* __restrict__ q_word_h,
                            const float* __restrict__ step_w1,
                            const float* __restrict__ step_b1,
                            const float* __restrict__ step_w2,
                            const float* __restrict__ step_b2,
                            const float* __restrict__ rel_w1,
                            const float* __restrict__ hop_w1,
                            const float* __restrict__ hop_b1,
                            const float* __restrict__ hop_w2,
                            const float* __restrict__ hop_b2,
                            _Float16* __restrict__ w1sw,
                            float* __restrict__ hop_attn)
{
    const int t   = blockIdx.x;
    const int tid = threadIdx.x;

    __shared__ float qe[DIM], h1[DIM], cq[DIM], ctx[DIM], qd[32], hh[256], hl[3];

    for (int c = tid; c < DIM; c += 256) qe[c] = q_emb[c];
    __syncthreads();

    const float* w1 = step_w1 + (long)t * DIM * DIM;
    const float* w2 = step_w2 + (long)t * DIM * DIM;

    for (int c = tid; c < DIM; c += 256) {
        float s = 0.f;
        for (int d = 0; d < DIM; ++d) s += qe[d] * w1[(long)d * DIM + c];
        h1[c] = gelu_exact(s + step_b1[t * DIM + c]);
    }
    __syncthreads();
    for (int c = tid; c < DIM; c += 256) {
        float s = 0.f;
        for (int d = 0; d < DIM; ++d) s += h1[d] * w2[(long)d * DIM + c];
        cq[c] = s + step_b2[t * DIM + c];
    }
    __syncthreads();
    if (tid < 32) {
        const float* qw = q_word_h + tid * DIM;
        float s = 0.f;
        for (int d = 0; d < DIM; ++d) s += cq[d] * qw[d];
        qd[tid] = s;
    }
    __syncthreads();
    if (tid == 0) {
        float m = qd[0];
        for (int l = 1; l < 32; ++l) m = fmaxf(m, qd[l]);
        float sum = 0.f;
        for (int l = 0; l < 32; ++l) { qd[l] = expf(qd[l] - m); sum += qd[l]; }
        float inv = 1.0f / sum;
        for (int l = 0; l < 32; ++l) qd[l] *= inv;
    }
    __syncthreads();
    for (int c = tid; c < DIM; c += 256) {
        float s = 0.f;
        for (int l = 0; l < 32; ++l) s += qd[l] * q_word_h[l * DIM + c];
        ctx[c] = s + cq[c];
    }
    __syncthreads();

    // W1t = ctx[:,None] * rel_w1, written pre-swizzled as f16 WMMA-B fragments
    for (int k = tid; k < DIM; k += 256) {
        const float cv = ctx[k];
        const int kc = k >> 5, kk = k & 31;
        const int lane_hi = (kk >> 4) << 4;
        const int p = kk & 15;
        const float* wrow = rel_w1 + (long)k * 256;
        for (int n = 0; n < 256; ++n) {
            const int lane = (n & 15) + lane_hi;
            const int nt = t * 16 + (n >> 4);
            w1sw[(((long)kc * 48 + nt) * 32 + lane) * 16 + p] = (_Float16)(cv * wrow[n]);
        }
    }

    if (t == 0) {
        {
            float s = 0.f;
            for (int d = 0; d < DIM; ++d) s += qe[d] * hop_w1[(long)d * 256 + tid];
            hh[tid] = gelu_exact(s + hop_b1[tid]);
        }
        __syncthreads();
        if (tid < 3) {
            float s = 0.f;
            for (int d = 0; d < 256; ++d) s += hh[d] * hop_w2[d * 3 + tid];
            hl[tid] = s + hop_b2[tid];
        }
        __syncthreads();
        if (tid == 0) {
            float m = fmaxf(hl[0], fmaxf(hl[1], hl[2]));
            float e0 = expf(hl[0] - m), e1 = expf(hl[1] - m), e2 = expf(hl[2] - m);
            float inv = 1.0f / (e0 + e1 + e2);
            hop_attn[0] = e0 * inv; hop_attn[1] = e1 * inv; hop_attn[2] = e2 * inv;
        }
    }
}

// ---------------------------------------------------------------------------
// Kernel 2: fused GEMM  dprob[t][e] = sigmoid(gelu(desc @ W1cat + b1) @ w2 + b2)
// 512 threads = 16 waves; wave (m in 0..1, n in 0..7) owns 2 M-tiles x 6 N-tiles.
// B chunk staged L2 -> LDS via GLOBAL_LOAD_ASYNC_TO_LDS_B128 (ASYNCcnt path).
// ---------------------------------------------------------------------------
__global__ __launch_bounds__(512) void gemm_kernel(
        const float* __restrict__ desc,
        const _Float16* __restrict__ w1sw,
        const float* __restrict__ rel_b1,
        const float* __restrict__ rel_w2,
        const float* __restrict__ rel_b2,
        float* __restrict__ dprob)
{
    __shared__ __align__(32) _Float16 Asw[4 * 32 * 16];    // 4 KB: 4 M-tiles of 16x32 f16
    __shared__ __align__(32) _Float16 Bsw[48 * 32 * 16];   // 48 KB: 48 N-tiles of 32x16 f16
    __shared__ float lg[3 * TM];

    const int tid    = threadIdx.x;
    const int lane   = tid & 31;
    const int wave   = tid >> 5;
    const int wave_m = wave >> 3;   // 0..1
    const int wave_n = wave & 7;    // 0..7
    const long row0  = (long)blockIdx.x * TM;

    // LDS byte address of Bsw for async-to-LDS (flat LDS aperture: low 32 bits)
    const unsigned bsw_lds = (unsigned)(size_t)(&Bsw[0]) + (unsigned)(tid * 96);
    const unsigned bsw_voff = (unsigned)(tid * 96);

    v8f acc[2][6];
    const v8f vzero = {};
#pragma unroll
    for (int a = 0; a < 2; ++a)
#pragma unroll
        for (int i = 0; i < 6; ++i) acc[a][i] = vzero;

    for (int kc = 0; kc < KCHUNKS; ++kc) {
        __syncthreads();
        // ---- stage B: async DMA 32x768 f16 chunk L2 -> LDS (no VGPR data) ----
        {
            const unsigned long sbase =
                (unsigned long)w1sw + (unsigned long)kc * 49152ul;
#pragma unroll
            for (int q = 0; q < 6; ++q) {
                asm volatile(
                    "global_load_async_to_lds_b128 %0, %1, %2"
                    :
                    : "v"(bsw_lds + q * 16), "v"(bsw_voff + q * 16), "s"(sbase)
                    : "memory");
            }
        }
        // ---- stage A: 64 rows x 32 K, f32 -> f16, WMMA-A fragment layout ----
        {
            const int r  = tid >> 3;          // 0..63
            const int kq = (tid & 7) * 4;     // 0,4,...,28
            const long row = row0 + r;
            float4 v = make_float4(0.f, 0.f, 0.f, 0.f);
            if (row < NUM_EDGES) {
                v = *(const float4*)(desc + row * DIM + kc * 32 + kq);
                if (kc + 1 < KCHUNKS)
                    __builtin_prefetch(desc + row * DIM + (kc + 1) * 32 + kq, 0, 1);
            }
            const float vv[4] = {v.x, v.y, v.z, v.w};
            const int mt = r >> 4, rl = r & 15;
#pragma unroll
            for (int e = 0; e < 4; ++e) {
                const int k   = kq + e;
                const int grp = k >> 3;                         // 0..3
                const int l2  = rl + ((grp >> 1) << 4);
                const int p   = k & 15;
                Asw[(mt * 32 + l2) * 16 + p] = (_Float16)vv[e];
            }
        }
        // retire async LDS writes, then workgroup barrier
        asm volatile("s_wait_asynccnt 0" ::: "memory");
        __syncthreads();
        // ---- 12 WMMAs per wave ----
        const v16h a0 = *(const v16h*)(Asw + ((wave_m * 2 + 0) * 32 + lane) * 16);
        const v16h a1 = *(const v16h*)(Asw + ((wave_m * 2 + 1) * 32 + lane) * 16);
#pragma unroll
        for (int i = 0; i < 6; ++i) {
            const v16h b = *(const v16h*)(Bsw + ((wave_n * 6 + i) * 32 + lane) * 16);
            acc[0][i] = __builtin_amdgcn_wmma_f32_16x16x32_f16(
                false, a0, false, b, (short)0, acc[0][i], false, false);
            acc[1][i] = __builtin_amdgcn_wmma_f32_16x16x32_f16(
                false, a1, false, b, (short)0, acc[1][i], false, false);
        }
    }

    // ---- fused epilogue: +b1, gelu, dot with rel_w2, +b2, sigmoid ----
    __syncthreads();
    if (tid < 3 * TM) lg[tid] = 0.0f;
    __syncthreads();

    const int c  = lane & 15;   // column within tile (C layout: col = lane&15)
    const int hb = lane >> 4;   // row half: rows v (hb=0) / 8+v (hb=1)
#pragma unroll
    for (int wm = 0; wm < 2; ++wm) {
#pragma unroll
        for (int i = 0; i < 6; ++i) {
            const int nt = wave_n * 6 + i;
            const int t  = nt >> 4;
            const int cb = ((nt & 15) << 4) + c;   // column within 256-block
            const float b1v = rel_b1[cb];
            const float w2v = rel_w2[cb];
#pragma unroll
            for (int v = 0; v < 8; ++v) {
                float s = gelu_exact(acc[wm][i][v] + b1v) * w2v;
                s += __shfl_xor(s, 1, 32);
                s += __shfl_xor(s, 2, 32);
                s += __shfl_xor(s, 4, 32);
                s += __shfl_xor(s, 8, 32);
                if (c == 0) {
                    const int rowl = (wave_m * 2 + wm) * 16 + v + (hb << 3);
                    atomicAdd(&lg[t * TM + rowl], s);
                }
            }
        }
    }
    __syncthreads();
    if (tid < 3 * TM) {
        const int t = tid >> 6, r = tid & (TM - 1);
        const long row = row0 + r;
        if (row < NUM_EDGES) {
            const float logit = lg[tid] + rel_b2[0];
            dprob[(long)t * NUM_EDGES + row] = 1.0f / (1.0f + expf(-logit));
        }
    }
}

// ---------------------------------------------------------------------------
// Propagation / reduction kernels
// ---------------------------------------------------------------------------
__global__ void zero_kernel(float* __restrict__ ebuf) {
    const long i = (long)blockIdx.x * blockDim.x + threadIdx.x;
    if (i < 4L * NUM_ENT) ebuf[i] = 0.0f;
}

__global__ void topic_kernel(float* __restrict__ e0, const int* __restrict__ topic) {
    if (threadIdx.x < 4) e0[topic[threadIdx.x]] = 1.0f;
}

__global__ void prop_kernel(const float* __restrict__ last_e,
                            const float* __restrict__ dprob_t,
                            const int* __restrict__ kb,
                            float* __restrict__ next_e) {
    const long i = (long)blockIdx.x * blockDim.x + threadIdx.x;
    if (i < NUM_EDGES) {
        const int sub = kb[2 * i], obj = kb[2 * i + 1];
        atomicAdd(&next_e[obj], last_e[sub] * dprob_t[i]);
    }
}

__global__ void norm_kernel(float* __restrict__ e) {
    const long i = (long)blockIdx.x * blockDim.x + threadIdx.x;
    if (i < NUM_ENT) {
        const float v = e[i];
        e[i] = v > 1.0f ? 1.0f : v;   // v / max(v,1) for v >= 0
    }
}

__global__ void final_kernel(const float* __restrict__ hop,
                             const float* __restrict__ e1,
                             const float* __restrict__ e2,
                             const float* __restrict__ e3,
                             float* __restrict__ out) {
    const long i = (long)blockIdx.x * blockDim.x + threadIdx.x;
    if (i < NUM_ENT) out[i] = hop[0] * e1[i] + hop[1] * e2[i] + hop[2] * e3[i];
}

// ---------------------------------------------------------------------------
extern "C" void kernel_launch(void* const* d_in, const int* in_sizes, int n_in,
                              void* d_out, int out_size, void* d_ws, size_t ws_size,
                              hipStream_t stream)
{
    const float* q_emb    = (const float*)d_in[0];
    const float* q_word_h = (const float*)d_in[1];
    const float* desc     = (const float*)d_in[2];
    const float* step_w1  = (const float*)d_in[3];
    const float* step_b1  = (const float*)d_in[4];
    const float* step_w2  = (const float*)d_in[5];
    const float* step_b2  = (const float*)d_in[6];
    const float* rel_w1   = (const float*)d_in[7];
    const float* rel_b1   = (const float*)d_in[8];
    const float* rel_w2   = (const float*)d_in[9];
    const float* rel_b2   = (const float*)d_in[10];
    const float* hop_w1   = (const float*)d_in[11];
    const float* hop_b1   = (const float*)d_in[12];
    const float* hop_w2   = (const float*)d_in[13];
    const float* hop_b2   = (const float*)d_in[14];
    const int*   kb_pair  = (const int*)d_in[15];
    const int*   topic    = (const int*)d_in[16];
    float* out = (float*)d_out;

    // workspace layout (bytes), all 16B-aligned: ~5.2 MB total
    char* ws = (char*)d_ws;
    _Float16* w1sw     = (_Float16*)(ws);                       // 3*768*256 f16 = 1,179,648 B
    float*    hop_attn = (float*)(ws + 1179648);                // 12 B (padded to 16)
    float*    dprob    = (float*)(ws + 1179664);                // 3*NUM_EDGES f32 = 2,400,000 B
    float*    ebuf     = (float*)(ws + 1179664 + 2400000);      // 4*NUM_ENT f32 = 1,600,000 B

    zero_kernel<<<(4 * NUM_ENT + 255) / 256, 256, 0, stream>>>(ebuf);
    topic_kernel<<<1, 32, 0, stream>>>(ebuf, topic);
    prep_kernel<<<3, 256, 0, stream>>>(q_emb, q_word_h, step_w1, step_b1, step_w2,
                                       step_b2, rel_w1, hop_w1, hop_b1, hop_w2,
                                       hop_b2, w1sw, hop_attn);
    gemm_kernel<<<(NUM_EDGES + TM - 1) / TM, 512, 0, stream>>>(desc, w1sw, rel_b1,
                                                               rel_w2, rel_b2, dprob);
    for (int t = 0; t < 3; ++t) {
        prop_kernel<<<(NUM_EDGES + 255) / 256, 256, 0, stream>>>(
            ebuf + (long)t * NUM_ENT, dprob + (long)t * NUM_EDGES, kb_pair,
            ebuf + (long)(t + 1) * NUM_ENT);
        norm_kernel<<<(NUM_ENT + 255) / 256, 256, 0, stream>>>(ebuf + (long)(t + 1) * NUM_ENT);
    }
    final_kernel<<<(NUM_ENT + 255) / 256, 256, 0, stream>>>(
        hop_attn, ebuf + NUM_ENT, ebuf + 2L * NUM_ENT, ebuf + 3L * NUM_ENT, out);
}
